// MambaFiLMDecoder_70720931496703
// MI455X (gfx1250) — compile-verified, matrix operand order checked by reading, not run on hardware
//
#include <hip/hip_runtime.h>
#include <math.h>

// ---------------------------------------------------------------------------
// MambaFiLMDecoder for MI455X (gfx1250, wave32).
// GEMMs: v_wmma_f32_16x16x32_bf16 (bf16 operands, f32 accum).
// Scan:  fp32 VALU sequential recurrence, 1 thread per (batch, channel).
// ---------------------------------------------------------------------------

#define DM    256
#define DI    512
#define DST   16
#define DTR   16
#define NL    4
#define BB    4
#define LL    2048
#define ROWS  (BB * LL)   // 8192

typedef __attribute__((ext_vector_type(16))) __bf16 v16bf;
typedef __attribute__((ext_vector_type(8)))  __bf16 v8bf;
typedef __attribute__((ext_vector_type(8)))  float  v8f;

// ------------------------- fp32 -> bf16 conversion -------------------------
__global__ void cvt_f32_bf16_kernel(const float* __restrict__ in,
                                    __bf16* __restrict__ out, int n) {
    int i = blockIdx.x * blockDim.x + threadIdx.x;
    if (i < n) out[i] = (__bf16)in[i];
}

// ------------------------- WMMA GEMM: C = A * W^T --------------------------
// A: [M,K] bf16 row-major.  W: [N,K] bf16 row-major (so B = W^T, col-major
// fetch of B == row fetch of W).  C: [M,N] f32.  All of M,N,K %16==0, K%32==0.
// One wave computes one 16x16 tile.
template <bool ADD_BIAS, bool RESIDUAL>
__global__ void gemm_bf16_wmma_kernel(const __bf16* __restrict__ A,
                                      const __bf16* __restrict__ W,
                                      const float* __restrict__ bias,
                                      const float* __restrict__ residual,
                                      float* __restrict__ C,
                                      int M, int N, int K) {
    const int lane  = threadIdx.x & 31;
    const int wave  = (blockIdx.x * blockDim.x + threadIdx.x) >> 5;
    const int tilesN = N >> 4;
    const int tm = wave / tilesN;
    const int tn = wave - tm * tilesN;
    if (tm * 16 >= M) return;

    const int hi  = lane >> 4;      // 0: K 0..7 / 16..23   1: K 8..15 / 24..31
    const int l16 = lane & 15;
    const __bf16* pA = A + (size_t)(tm * 16 + l16) * K + hi * 8;
    const __bf16* pW = W + (size_t)(tn * 16 + l16) * K + hi * 8;

    v8f acc = {};
    for (int k0 = 0; k0 < K; k0 += 32) {
        v8bf alo = *(const v8bf*)(pA + k0);
        v8bf ahi = *(const v8bf*)(pA + k0 + 16);
        v8bf blo = *(const v8bf*)(pW + k0);
        v8bf bhi = *(const v8bf*)(pW + k0 + 16);
        __builtin_prefetch(pA + k0 + 128, 0, 1);   // global_prefetch_b8
        __builtin_prefetch(pW + k0 + 128, 0, 1);
        v16bf av, bv;
#pragma unroll
        for (int i = 0; i < 8; ++i) {
            av[i] = alo[i]; av[8 + i] = ahi[i];
            bv[i] = blo[i]; bv[8 + i] = bhi[i];
        }
        acc = __builtin_amdgcn_wmma_f32_16x16x32_bf16(
            false, av, false, bv, (short)0, acc, false, false);
    }

    const int col   = tn * 16 + l16;
    const int rbase = tm * 16 + hi * 8;   // C layout: vgpr v -> M = v (+8 hi)
#pragma unroll
    for (int v = 0; v < 8; ++v) {
        float val = acc[v];
        if (ADD_BIAS) val += bias[col];
        size_t idx = (size_t)(rbase + v) * N + col;
        if (RESIDUAL) val += residual[idx];
        C[idx] = val;
    }
}

// ------------------------- context projection (tiny) -----------------------
__global__ void ctx_proj_kernel(const float* __restrict__ context,   // [4,768]
                                const float* __restrict__ w,         // [256,768]
                                const float* __restrict__ b,         // [256]
                                float* __restrict__ ctx) {           // [4,256]
    int d = blockIdx.x * blockDim.x + threadIdx.x;
    if (d >= DM) return;
    for (int bb = 0; bb < BB; ++bb) {
        float a = b[d];
        const float* wr = w + (size_t)d * 768;
        const float* cr = context + (size_t)bb * 768;
        for (int k = 0; k < 768; ++k) a = fmaf(cr[k], wr[k], a);
        ctx[bb * DM + d] = a;
    }
}

// ------------------------- FiLM parameters (tiny) --------------------------
__global__ void film_kernel(const float* __restrict__ ctx,      // [4,256]
                            const float* __restrict__ film_w,   // [NL,512,256]
                            const float* __restrict__ film_b,   // [NL,512]
                            float* __restrict__ film) {         // [NL,4,512]
    int layer = blockIdx.x >> 2;
    int bb    = blockIdx.x & 3;
    int j     = threadIdx.x;                                    // 0..511
    const float* w = film_w + ((size_t)layer * 2 * DM + j) * DM;
    const float* c = ctx + bb * DM;
    float a = film_b[layer * 2 * DM + j];
    for (int d = 0; d < DM; ++d) a = fmaf(c[d], w[d], a);
    film[((size_t)layer * BB + bb) * 2 * DM + j] = a;
}

// -------------------- RMSNorm + FiLM modulation -> bf16 ---------------------
__global__ void rms_film_kernel(const float* __restrict__ h,       // [ROWS,256]
                                const float* __restrict__ norm_w,  // [256]
                                const float* __restrict__ film,    // [4,512]
                                __bf16* __restrict__ xin) {        // [ROWS,256]
    int row = blockIdx.x;
    int bb  = row / LL;
    int d   = threadIdx.x;                                         // 256 thr
    float v = h[(size_t)row * DM + d];
    __shared__ float s[8];
    float ss = v * v;
#pragma unroll
    for (int o = 16; o > 0; o >>= 1) ss += __shfl_down(ss, o, 32);
    if ((threadIdx.x & 31) == 0) s[threadIdx.x >> 5] = ss;
    __syncthreads();
    if (threadIdx.x == 0) {
        float t = 0.f;
#pragma unroll
        for (int i = 0; i < 8; ++i) t += s[i];
        s[0] = rsqrtf(t * (1.0f / DM) + 1e-5f);
    }
    __syncthreads();
    float inv  = s[0];
    float g    = film[bb * 2 * DM + d];
    float beta = film[bb * 2 * DM + DM + d];
    xin[(size_t)row * DM + d] = (__bf16)(v * inv * norm_w[d] * (1.0f + g) + beta);
}

// ---------------- causal depthwise conv (D_CONV=4) + SiLU -------------------
__global__ void conv_silu_kernel(const float* __restrict__ xz,     // [ROWS,1024]
                                 const float* __restrict__ conv_w, // [512,4]
                                 const float* __restrict__ conv_b, // [512]
                                 float* __restrict__ xc,           // [ROWS,512]
                                 __bf16* __restrict__ xc_bf) {
    int idx = blockIdx.x * blockDim.x + threadIdx.x;
    if (idx >= ROWS * DI) return;
    int e   = idx & (DI - 1);
    int row = idx >> 9;
    int l   = row & (LL - 1);
    float a = conv_b[e];
#pragma unroll
    for (int k = 0; k < 4; ++k) {
        int lk = l - 3 + k;
        if (lk >= 0)
            a = fmaf(conv_w[e * 4 + k], xz[(size_t)(row - 3 + k) * (2 * DI) + e], a);
    }
    float s = a / (1.0f + __expf(-a));      // SiLU
    xc[idx]    = s;
    xc_bf[idx] = (__bf16)s;
}

// ------------------- delta = softplus(dr @ dt_w^T + dt_b) -------------------
__global__ void delta_kernel(const float* __restrict__ dbc,   // [ROWS,48]
                             const float* __restrict__ dt_w,  // [512,16]
                             const float* __restrict__ dt_b,  // [512]
                             float* __restrict__ delta) {
    int idx = blockIdx.x * blockDim.x + threadIdx.x;
    if (idx >= ROWS * DI) return;
    int e   = idx & (DI - 1);
    int row = idx >> 9;
    const float* dr = dbc + (size_t)row * 48;
    float a = dt_b[e];
#pragma unroll
    for (int r = 0; r < DTR; ++r) a = fmaf(dr[r], dt_w[e * DTR + r], a);
    delta[idx] = (a > 20.f) ? a : log1pf(__expf(a));
}

// ---------------- selective scan + gate; 1 thread per (b,e) -----------------
__global__ void scan_kernel(const float* __restrict__ delta,  // [ROWS,512]
                            const float* __restrict__ xc,     // [ROWS,512]
                            const float* __restrict__ xz,     // [ROWS,1024]
                            const float* __restrict__ dbc,    // [ROWS,48]
                            const float* __restrict__ A_log,  // [512,16]
                            const float* __restrict__ Dp,     // [512]
                            __bf16* __restrict__ yz) {        // [ROWS,512]
    int t = blockIdx.x * blockDim.x + threadIdx.x;
    if (t >= BB * DI) return;
    int e  = t & (DI - 1);
    int bb = t >> 9;
    float Ae[DST], st[DST];
#pragma unroll
    for (int n = 0; n < DST; ++n) {
        Ae[n] = -__expf(A_log[e * DST + n]);
        st[n] = 0.f;
    }
    float dpe = Dp[e];
    size_t base = (size_t)bb * LL;
    for (int l = 0; l < LL; ++l) {
        size_t row = base + l;
        float d  = delta[row * DI + e];
        float xv = xc[row * DI + e];
        float zv = xz[row * (2 * DI) + DI + e];
        const float* bc = dbc + row * 48 + DTR;   // [0..15]=B, [16..31]=C
        float dx = d * xv;
        float y  = dpe * xv;
#pragma unroll
        for (int n = 0; n < DST; ++n) {
            float da = __expf(d * Ae[n]);
            st[n] = fmaf(da, st[n], dx * bc[n]);
            y = fmaf(st[n], bc[DST + n], y);
        }
        y *= zv / (1.0f + __expf(-zv));           // * silu(z)
        yz[row * DI + e] = (__bf16)y;
    }
}

// ------------------------- final pooling ------------------------------------
__global__ void zero_kernel(float* __restrict__ p, int n) {
    int i = blockIdx.x * blockDim.x + threadIdx.x;
    if (i < n) p[i] = 0.f;
}

__global__ void pool_kernel(const float* __restrict__ h,      // [ROWS,256]
                            const float* __restrict__ normf,  // [256]
                            float* __restrict__ pooled) {     // [4,256]
    int row = blockIdx.x;
    int bb  = row / LL;
    int d   = threadIdx.x;
    float v = h[(size_t)row * DM + d];
    __shared__ float s[8];
    float ss = v * v;
#pragma unroll
    for (int o = 16; o > 0; o >>= 1) ss += __shfl_down(ss, o, 32);
    if ((threadIdx.x & 31) == 0) s[threadIdx.x >> 5] = ss;
    __syncthreads();
    if (threadIdx.x == 0) {
        float t = 0.f;
#pragma unroll
        for (int i = 0; i < 8; ++i) t += s[i];
        s[0] = rsqrtf(t * (1.0f / DM) + 1e-5f);
    }
    __syncthreads();
    atomicAdd(&pooled[bb * DM + d], v * s[0] * normf[d] * (1.0f / LL));
}

// ------------------------- classifier head (tiny) ---------------------------
__global__ void cls_kernel(const float* __restrict__ pooled,  // [4,256]
                           const float* __restrict__ w1,      // [512,256]
                           const float* __restrict__ b1,      // [512]
                           const float* __restrict__ w2,      // [4,512]
                           const float* __restrict__ b2,      // [4]
                           float* __restrict__ out) {         // [4,4]
    __shared__ float hid[BB * 2 * DM];
    int j = threadIdx.x;                                      // 512 threads
    for (int bb = 0; bb < BB; ++bb) {
        float a = b1[j];
        const float* wr = w1 + (size_t)j * DM;
        const float* pr = pooled + bb * DM;
        for (int d = 0; d < DM; ++d) a = fmaf(pr[d], wr[d], a);
        hid[bb * 2 * DM + j] = fmaxf(a, 0.f);
    }
    __syncthreads();
    if (j < 16) {
        int bb = j >> 2, c = j & 3;
        float a = b2[c];
        for (int k = 0; k < 2 * DM; ++k)
            a = fmaf(hid[bb * 2 * DM + k], w2[c * 2 * DM + k], a);
        out[bb * 4 + c] = a;
    }
}

// ---------------------------------------------------------------------------
static inline int cdiv(int a, int b) { return (a + b - 1) / b; }

extern "C" void kernel_launch(void* const* d_in, const int* in_sizes, int n_in,
                              void* d_out, int out_size, void* d_ws, size_t ws_size,
                              hipStream_t stream) {
    const float* x       = (const float*)d_in[0];   // [4,2048,768]
    const float* context = (const float*)d_in[1];   // [4,768]
    const float* ipw     = (const float*)d_in[2];   // [256,768]
    const float* ipb     = (const float*)d_in[3];
    const float* cpw     = (const float*)d_in[4];   // [256,768]
    const float* cpb     = (const float*)d_in[5];
    const float* norm_w  = (const float*)d_in[6];   // [4,256]
    const float* film_w  = (const float*)d_in[7];   // [4,512,256]
    const float* film_b  = (const float*)d_in[8];   // [4,512]
    const float* in_w    = (const float*)d_in[9];   // [4,1024,256]
    const float* conv_w  = (const float*)d_in[10];  // [4,512,4]
    const float* conv_b  = (const float*)d_in[11];  // [4,512]
    const float* xproj_w = (const float*)d_in[12];  // [4,48,512]
    const float* dt_w    = (const float*)d_in[13];  // [4,512,16]
    const float* dt_b    = (const float*)d_in[14];  // [4,512]
    const float* A_log   = (const float*)d_in[15];  // [4,512,16]
    const float* Dp      = (const float*)d_in[16];  // [4,512]
    const float* out_w   = (const float*)d_in[17];  // [4,256,512]
    const float* normf_w = (const float*)d_in[18];  // [256]
    const float* cls1_w  = (const float*)d_in[19];  // [512,256]
    const float* cls1_b  = (const float*)d_in[20];
    const float* cls2_w  = (const float*)d_in[21];  // [4,512]
    const float* cls2_b  = (const float*)d_in[22];
    float* out = (float*)d_out;                     // [4,4]

    // ---- workspace carve-up (256B aligned) ----
    char* base = (char*)d_ws;
    size_t off = 0;
    auto take = [&](size_t bytes) {
        off = (off + 255) & ~(size_t)255;
        char* p = base + off;
        off += bytes;
        return p;
    };
    __bf16* x_bf    = (__bf16*)take((size_t)BB * LL * 768 * 2);
    __bf16* ipw_bf  = (__bf16*)take((size_t)DM * 768 * 2);
    __bf16* inw_bf  = (__bf16*)take((size_t)NL * 2 * DI * DM * 2);
    __bf16* xpw_bf  = (__bf16*)take((size_t)NL * 48 * DI * 2);
    __bf16* outw_bf = (__bf16*)take((size_t)NL * DM * DI * 2);
    float*  h       = (float*)take((size_t)ROWS * DM * 4);
    __bf16* xin_bf  = (__bf16*)take((size_t)ROWS * DM * 2);
    float*  xzb     = (float*)take((size_t)ROWS * 2 * DI * 4);
    float*  xc      = (float*)take((size_t)ROWS * DI * 4);
    __bf16* xc_bf   = (__bf16*)take((size_t)ROWS * DI * 2);
    float*  dbc     = (float*)take((size_t)ROWS * 48 * 4);
    float*  delta   = (float*)take((size_t)ROWS * DI * 4);
    __bf16* yz_bf   = (__bf16*)take((size_t)ROWS * DI * 2);
    float*  ctx     = (float*)take((size_t)BB * DM * 4);
    float*  film    = (float*)take((size_t)NL * BB * 2 * DM * 4);
    float*  pooled  = (float*)take((size_t)BB * DM * 4);
    (void)ws_size; (void)n_in; (void)in_sizes; (void)out_size;

    // ---- bf16 conversions (activations + GEMM weights) ----
    auto cvt = [&](const float* src, __bf16* dst, int n) {
        cvt_f32_bf16_kernel<<<cdiv(n, 256), 256, 0, stream>>>(src, dst, n);
    };
    cvt(x,       x_bf,    BB * LL * 768);
    cvt(ipw,     ipw_bf,  DM * 768);
    cvt(in_w,    inw_bf,  NL * 2 * DI * DM);
    cvt(xproj_w, xpw_bf,  NL * 48 * DI);
    cvt(out_w,   outw_bf, NL * DM * DI);

    // ---- context / FiLM ----
    ctx_proj_kernel<<<1, 256, 0, stream>>>(context, cpw, cpb, ctx);
    film_kernel<<<NL * BB, 2 * DM, 0, stream>>>(ctx, film_w, film_b, film);

    auto gemm_blocks = [&](int M, int N) { return cdiv((M / 16) * (N / 16) * 32, 256); };

    // ---- h = x @ input_proj_w^T + b ----
    gemm_bf16_wmma_kernel<true, false><<<gemm_blocks(ROWS, DM), 256, 0, stream>>>(
        x_bf, ipw_bf, ipb, nullptr, h, ROWS, DM, 768);

    // ---- layers ----
    for (int layer = 0; layer < NL; ++layer) {
        rms_film_kernel<<<ROWS, DM, 0, stream>>>(
            h, norm_w + layer * DM, film + (size_t)layer * BB * 2 * DM, xin_bf);

        gemm_bf16_wmma_kernel<false, false><<<gemm_blocks(ROWS, 2 * DI), 256, 0, stream>>>(
            xin_bf, inw_bf + (size_t)layer * 2 * DI * DM, nullptr, nullptr,
            xzb, ROWS, 2 * DI, DM);

        conv_silu_kernel<<<cdiv(ROWS * DI, 256), 256, 0, stream>>>(
            xzb, conv_w + layer * DI * 4, conv_b + layer * DI, xc, xc_bf);

        gemm_bf16_wmma_kernel<false, false><<<gemm_blocks(ROWS, 48), 256, 0, stream>>>(
            xc_bf, xpw_bf + (size_t)layer * 48 * DI, nullptr, nullptr,
            dbc, ROWS, 48, DI);

        delta_kernel<<<cdiv(ROWS * DI, 256), 256, 0, stream>>>(
            dbc, dt_w + layer * DI * DTR, dt_b + layer * DI, delta);

        scan_kernel<<<cdiv(BB * DI, 256), 256, 0, stream>>>(
            delta, xc, xzb, dbc, A_log + layer * DI * DST, Dp + layer * DI, yz_bf);

        gemm_bf16_wmma_kernel<false, true><<<gemm_blocks(ROWS, DM), 256, 0, stream>>>(
            yz_bf, outw_bf + (size_t)layer * DM * DI, nullptr, h /*residual*/,
            h, ROWS, DM, DI);
    }

    // ---- pooled = mean_l rms(h)*normf ----
    zero_kernel<<<cdiv(BB * DM, 256), 256, 0, stream>>>(pooled, BB * DM);
    pool_kernel<<<ROWS, DM, 0, stream>>>(h, normf_w, pooled);

    // ---- classifier ----
    cls_kernel<<<1, 2 * DM, 0, stream>>>(pooled, cls1_w, cls1_b, cls2_w, cls2_b, out);
}
